// HierarchicalSoftMoE2DBlock_27255862460951
// MI455X (gfx1250) — matrix operand order repaired and checked by reading, not compiled, not run on hardware
//
#include <hip/hip_runtime.h>
#include <math.h>

// ---------------- problem constants (match reference) ----------------
#define Bb   4
#define Nn   4096
#define Dd   768
#define Gg   16            // groups per batch
#define BG   (Bb * Gg)     // 64 total groups
#define GRP  256           // tokens per group
#define E1c  8
#define S1c  16
#define E2c  4
#define Mm   4             // m = S1/ns
#define P1   (E1c * S1c)   // 128
#define HIDc 2048
#define SCALE 27.712812921102035f   // sqrt(768)
#define TS   36            // LDS row stride (floats) for transposed weight tiles

// ---------------- WMMA helpers (gfx1250, wave32) ----------------
typedef __attribute__((ext_vector_type(16))) __bf16 v16bf;
typedef __attribute__((ext_vector_type(8)))  float  v8f;

__device__ __forceinline__ v8f wmma_bf16(v16bf a, v16bf b, v8f c) {
  // 8 args: (neg_a, A, neg_b, B, c_mod, C, reuse_a, reuse_b)
  return __builtin_amdgcn_wmma_f32_16x16x32_bf16(false, a, false, b, (short)0, c,
                                                 false, false);
}

// A fragment 16x32 (MxK), row-major source; rowp -> element (row_of_this_lane, k0)
// K order used here (contiguous, B matches): lanes 0-15 hold K 0-15 ; lanes 16-31 hold K 16-31.
__device__ __forceinline__ v16bf a_frag(const float* rowp, int lane) {
  const int lo = (lane < 16) ? 0 : 16;
  const int hi = (lane < 16) ? 8 : 24;
  v16bf f;
#pragma unroll
  for (int e = 0; e < 8; ++e) f[e] = (__bf16)rowp[lo + e];
#pragma unroll
  for (int e = 0; e < 8; ++e) f[8 + e] = (__bf16)rowp[hi + e];
  return f;
}

// A fragment where the logical A(MxK) is stored K-major (transposed, ld = ldm):
// p -> element (k0, row_of_this_lane)
__device__ __forceinline__ v16bf at_frag(const float* p, int ldm, int lane) {
  const int lo = (lane < 16) ? 0 : 16;
  const int hi = (lane < 16) ? 8 : 24;
  v16bf f;
#pragma unroll
  for (int e = 0; e < 8; ++e) f[e] = (__bf16)p[(size_t)(lo + e) * ldm];
#pragma unroll
  for (int e = 0; e < 8; ++e) f[8 + e] = (__bf16)p[(size_t)(hi + e) * ldm];
  return f;
}

// B fragment 32x16 (KxN), row-major source; p -> element (k0 + (lane<16?0:16), col)
__device__ __forceinline__ v16bf b_frag(const float* p, int ldb) {
  v16bf f;
#pragma unroll
  for (int e = 0; e < 16; ++e) f[e] = (__bf16)p[(size_t)e * ldb];
  return f;
}

// B fragment where logical B(KxN) is stored NxK (contiguous in K):
// p -> element (col, k0 + (lane<16?0:16))
__device__ __forceinline__ v16bf bt_frag(const float* p) {
  v16bf f;
#pragma unroll
  for (int e = 0; e < 16; ++e) f[e] = (__bf16)p[e];
  return f;
}

// C/D f32 16x16: VGPR r, lanes 0-15 -> M=r, lanes 16-31 -> M=r+8, N=lane&15
__device__ __forceinline__ void store_c(float* base, int ldc, int row0, int col0,
                                        int lane, v8f acc) {
  const int c  = col0 + (lane & 15);
  const int r0 = row0 + ((lane < 16) ? 0 : 8);
#pragma unroll
  for (int r = 0; r < 8; ++r) base[(size_t)(r0 + r) * ldc + c] = acc[r];
}

// ---------------- small helpers ----------------
__device__ __forceinline__ float wave_sum(float v) {
#pragma unroll
  for (int o = 16; o > 0; o >>= 1) v += __shfl_xor(v, o, 32);
  return v;
}
__device__ __forceinline__ float wave_max(float v) {
#pragma unroll
  for (int o = 16; o > 0; o >>= 1) v = fmaxf(v, __shfl_xor(v, o, 32));
  return v;
}
__device__ __forceinline__ float gelu_tanh(float x) {   // jax.nn.gelu (approximate)
  const float x3 = x * x * x;
  return 0.5f * x * (1.0f + tanhf(0.7978845608028654f * (x + 0.044715f * x3)));
}

// ================= K0: RMSNorm of x -> xn =================
__global__ void k_rmsnorm(const float* __restrict__ x, const float* __restrict__ gamma,
                          float* __restrict__ xn) {
  const size_t t = blockIdx.x;
  const float* xi = x + t * Dd;
  float* xo = xn + t * Dd;
  __shared__ float red[8];
  float s = 0.f;
  for (int i = threadIdx.x; i < Dd; i += 256) { const float v = xi[i]; s += v * v; }
  s = wave_sum(s);
  if ((threadIdx.x & 31) == 0) red[threadIdx.x >> 5] = s;
  __syncthreads();
  float tot = 0.f;
#pragma unroll
  for (int j = 0; j < 8; ++j) tot += red[j];
  const float rs = rsqrtf(tot + 1e-6f) * SCALE;
  for (int i = threadIdx.x; i < Dd; i += 256) xo[i] = xi[i] * rs * gamma[i];
}

// ================= K0b: phi = l2norm(slots1) * sqrt(d) =================
__global__ void k_phinorm(const float* __restrict__ slots1, float* __restrict__ phi) {
  const size_t t = blockIdx.x;   // 0..127 (e*16+s)
  const float* xi = slots1 + t * Dd;
  float* xo = phi + t * Dd;
  __shared__ float red[8];
  float s = 0.f;
  for (int i = threadIdx.x; i < Dd; i += 256) { const float v = xi[i]; s += v * v; }
  s = wave_sum(s);
  if ((threadIdx.x & 31) == 0) red[threadIdx.x >> 5] = s;
  __syncthreads();
  float tot = 0.f;
#pragma unroll
  for (int j = 0; j < 8; ++j) tot += red[j];
  const float rs = rsqrtf(tot + 1e-6f) * SCALE;
  for (int i = threadIdx.x; i < Dd; i += 256) xo[i] = xi[i] * rs;
}

// ================= K1a: logits1 = xg (256x768) * phi^T (768x128) =================
__global__ void k1a_logits1(const float* __restrict__ xn, const float* __restrict__ phi,
                            float* __restrict__ logits1) {
  const int bg = blockIdx.x;
  const float* Ax = xn + (size_t)bg * GRP * Dd;
  float* Cc = logits1 + (size_t)bg * GRP * P1;
  const int lane = threadIdx.x & 31, w = threadIdx.x >> 5;
#pragma unroll 1
  for (int i = 0; i < 16; ++i) {                 // 16x8 = 128 tiles / 8 waves
    const int tidx = w * 16 + i;
    const int mt = tidx >> 3, nt = tidx & 7;
    const int row = mt * 16 + (lane & 15);
    const int col = nt * 16 + (lane & 15);
    v8f acc = {};
    for (int kt = 0; kt < Dd / 32; ++kt) {
      v16bf a = a_frag(Ax + (size_t)row * Dd + kt * 32, lane);
      v16bf b = bt_frag(phi + (size_t)col * Dd + kt * 32 + ((lane < 16) ? 0 : 16));
      acc = wmma_bf16(a, b, acc);
    }
    store_c(Cc, P1, mt * 16, nt * 16, lane, acc);
  }
}

// ================= K1b: comb1 = softmax over the 128 (e,s) per token ============
__global__ void k1b_comb1(const float* __restrict__ logits1, float* __restrict__ comb1) {
  const int token = blockIdx.x * 8 + (threadIdx.x >> 5);
  const int lane = threadIdx.x & 31;
  const float* L = logits1 + (size_t)token * P1;
  float v[4];
  float mx = -3.0e38f;
#pragma unroll
  for (int j = 0; j < 4; ++j) { v[j] = L[lane + 32 * j]; mx = fmaxf(mx, v[j]); }
  mx = wave_max(mx);
  float sum = 0.f;
#pragma unroll
  for (int j = 0; j < 4; ++j) { v[j] = __expf(v[j] - mx); sum += v[j]; }
  sum = wave_sum(sum);
  const float inv = 1.f / sum;
  float* Cm = comb1 + (size_t)token * P1;
#pragma unroll
  for (int j = 0; j < 4; ++j) Cm[lane + 32 * j] = v[j] * inv;
}

// ============ K1c: disp1 = softmax over 256 tokens per (e,s), in-place ==========
__global__ void k1c_disp1(float* __restrict__ logits1) {
  const int bg = blockIdx.x;
  float* L = logits1 + (size_t)bg * GRP * P1;
  const int c = threadIdx.x;       // 128 threads, one column each
  float mx = -3.0e38f;
  for (int r = 0; r < GRP; ++r) mx = fmaxf(mx, L[(size_t)r * P1 + c]);
  float sum = 0.f;
  for (int r = 0; r < GRP; ++r) sum += __expf(L[(size_t)r * P1 + c] - mx);
  const float inv = 1.f / sum;
  for (int r = 0; r < GRP; ++r)
    L[(size_t)r * P1 + c] = __expf(L[(size_t)r * P1 + c] - mx) * inv;
}

// ======== K1d: slots(128x768) = disp1^T (128x256) * xg (256x768) (WMMA) =========
__global__ void k1d_slots(const float* __restrict__ disp1, const float* __restrict__ xn,
                          float* __restrict__ slots) {
  const int bg = blockIdx.x;
  const float* L  = disp1 + (size_t)bg * GRP * P1;   // stored (token, p): K-major for A
  const float* Bx = xn + (size_t)bg * GRP * Dd;
  float* Cs = slots + (size_t)bg * P1 * Dd;
  const int lane = threadIdx.x & 31, w = threadIdx.x >> 5;
#pragma unroll 1
  for (int i = 0; i < 48; ++i) {                 // 8x48 = 384 tiles / 8 waves
    const int tidx = w * 48 + i;
    const int mt = tidx / 48, nt = tidx % 48;
    const int prow = mt * 16 + (lane & 15);
    const int col  = nt * 16 + (lane & 15);
    v8f acc = {};
    for (int kt = 0; kt < GRP / 32; ++kt) {
      v16bf a = at_frag(L + (size_t)(kt * 32) * P1 + prow, P1, lane);
      v16bf b = b_frag(Bx + (size_t)(kt * 32 + ((lane < 16) ? 0 : 16)) * Dd + col, Dd);
      acc = wmma_bf16(a, b, acc);
    }
    store_c(Cs, Dd, mt * 16, nt * 16, lane, acc);
  }
}

// ============ K2a: yn = l2norm(slots)*scale*gamma2 ; segment means ==============
__global__ void k2a_norm(const float* __restrict__ slots, const float* __restrict__ gamma2,
                         float* __restrict__ ynb, float* __restrict__ meansb) {
  const int bg = blockIdx.x / E1c, e1 = blockIdx.x % E1c;
  const float* S = slots + ((size_t)bg * P1 + e1 * S1c) * Dd;
  __shared__ float ynl[S1c * Dd];        // 48 KB
  __shared__ float part[16][17];
  __shared__ float rsn[16];
  const int tid = threadIdx.x;
  {
    const int r = tid >> 4, c16 = tid & 15;
    float s = 0.f;
    for (int c = c16; c < Dd; c += 16) { const float v = S[(size_t)r * Dd + c]; s += v * v; }
    part[r][c16] = s;
  }
  __syncthreads();
  if (tid < 16) {
    float t = 0.f;
#pragma unroll
    for (int j = 0; j < 16; ++j) t += part[tid][j];
    rsn[tid] = rsqrtf(t + 1e-6f) * SCALE;
  }
  __syncthreads();
  const float* gm = gamma2 + (size_t)e1 * Dd;
  float* yo = ynb + ((size_t)bg * E1c + e1) * S1c * Dd;
  for (int i = tid; i < S1c * Dd; i += 256) {
    const int rr = i / Dd, cc = i % Dd;
    const float v = S[i] * rsn[rr] * gm[cc];
    ynl[i] = v;
    yo[i]  = v;
  }
  __syncthreads();
  float* mo = meansb + ((size_t)bg * E1c + e1) * Mm * Dd;
  for (int i = tid; i < Mm * Dd; i += 256) {
    const int mm = i / Dd, cc = i % Dd;
    mo[i] = 0.25f * (ynl[(mm * 4 + 0) * Dd + cc] + ynl[(mm * 4 + 1) * Dd + cc] +
                     ynl[(mm * 4 + 2) * Dd + cc] + ynl[(mm * 4 + 3) * Dd + cc]);
  }
}

// ==== K2b: se2raw = means_batched (256x768) * w_slot2[e1] (768x3072) (WMMA) =====
// batched over all (b,G) so w_slot2 streams from HBM exactly once
__global__ void k2b_se2(const float* __restrict__ meansb, const float* __restrict__ w_slot2,
                        float* __restrict__ se2raw) {
  const int e1 = blockIdx.x / 24;
  const int c0 = (blockIdx.x % 24) * 128;
  const int lane = threadIdx.x & 31, w = threadIdx.x >> 5;
  const float* Bw = w_slot2 + (size_t)e1 * Dd * (E2c * Dd);
  const int ldb = E2c * Dd;   // 3072
#pragma unroll 1
  for (int i = 0; i < 16; ++i) {                 // 16x8 tiles / 8 waves
    const int tidx = w * 16 + i;
    const int mt = tidx >> 3, nt = tidx & 7;
    const int row = mt * 16 + (lane & 15);       // row = bg*4 + mm
    const int col = c0 + nt * 16 + (lane & 15);
    const float* arow =
        meansb + (((size_t)(row >> 2) * E1c + e1) * Mm + (row & 3)) * Dd;
    v8f acc = {};
    for (int kt = 0; kt < Dd / 32; ++kt) {
      const float* bp = Bw + (size_t)(kt * 32 + ((lane < 16) ? 0 : 16)) * ldb + col;
      __builtin_prefetch(bp + 32 * (size_t)ldb, 0, 0);
      v16bf a = a_frag(arow + kt * 32, lane);
      acc = wmma_bf16(a, b_frag(bp, ldb), acc);
    }
    const int r0 = mt * 16 + ((lane < 16) ? 0 : 8);
#pragma unroll
    for (int r = 0; r < 8; ++r) {
      const int rr = r0 + r;
      se2raw[(((size_t)(rr >> 2) * E1c + e1) * Mm + (rr & 3)) * (size_t)ldb + col] = acc[r];
    }
  }
}

// ====== K2c: normalize se2, 16x16 routing, disp2/comb2 softmax, slots2 ==========
__global__ void k2c_route(const float* __restrict__ ynb, const float* __restrict__ se2raw,
                          float* __restrict__ comb2, float* __restrict__ slots2) {
  const int bg = blockIdx.x / E1c, e1 = blockIdx.x % E1c;
  const float* yn = ynb + ((size_t)bg * E1c + e1) * S1c * Dd;
  const float* sr = se2raw + ((size_t)bg * E1c + e1) * Mm * (size_t)(E2c * Dd);
  __shared__ float se[S1c * Dd];         // 48 KB: normalized se2, rows em = e*4+mm
  __shared__ float part[16][17];
  __shared__ float rsn[16];
  __shared__ float lg[16][16];
  __shared__ float dp[16][16];
  const int tid = threadIdx.x;
  {
    const int r = tid >> 4, c16 = tid & 15;    // r = em
    const float* v = sr + (size_t)(r & 3) * (E2c * Dd) + (size_t)(r >> 2) * Dd;
    float s = 0.f;
    for (int c = c16; c < Dd; c += 16) s += v[c] * v[c];
    part[r][c16] = s;
  }
  __syncthreads();
  if (tid < 16) {
    float t = 0.f;
#pragma unroll
    for (int j = 0; j < 16; ++j) t += part[tid][j];
    rsn[tid] = rsqrtf(t + 1e-6f) * SCALE;
  }
  __syncthreads();
  for (int i = tid; i < S1c * Dd; i += 256) {
    const int rr = i / Dd, cc = i % Dd;
    se[i] = sr[(size_t)(rr & 3) * (E2c * Dd) + (size_t)(rr >> 2) * Dd + cc] * rsn[rr];
  }
  __syncthreads();
  {  // logits2[s][em] = dot(yn[s], se[em])
    const int s = tid >> 4, em = tid & 15;
    float acc = 0.f;
    const float* a = yn + (size_t)s * Dd;
    const float* b = se + (size_t)em * Dd;
    for (int d = 0; d < Dd; ++d) acc += a[d] * b[d];
    lg[s][em] = acc;
  }
  __syncthreads();
  if (tid < 16) {                      // disp2: softmax over s per em
    const int em = tid;
    float mx = -3.0e38f;
    for (int s = 0; s < 16; ++s) mx = fmaxf(mx, lg[s][em]);
    float su = 0.f;
    for (int s = 0; s < 16; ++s) { const float e = __expf(lg[s][em] - mx); dp[s][em] = e; su += e; }
    const float inv = 1.f / su;
    for (int s = 0; s < 16; ++s) dp[s][em] *= inv;
  }
  if (tid >= 16 && tid < 32) {         // comb2: softmax over em per s -> global
    const int s = tid - 16;
    float mx = -3.0e38f;
    for (int em = 0; em < 16; ++em) mx = fmaxf(mx, lg[s][em]);
    float su = 0.f;
    float tmp[16];
    for (int em = 0; em < 16; ++em) { tmp[em] = __expf(lg[s][em] - mx); su += tmp[em]; }
    const float inv = 1.f / su;
    float* c2 = comb2 + (((size_t)bg * E1c + e1) * S1c + s) * S1c;
    for (int em = 0; em < 16; ++em) c2[em] = tmp[em] * inv;
  }
  __syncthreads();
  float* so = slots2 + ((size_t)bg * E1c + e1) * S1c * Dd;   // rows em = e2*4+mm
  for (int i = tid; i < S1c * Dd; i += 256) {
    const int em = i / Dd, d = i % Dd;
    float acc = 0.f;
#pragma unroll
    for (int s = 0; s < 16; ++s) acc += dp[s][em] * yn[(size_t)s * Dd + d];
    so[i] = acc;
  }
}

// ====== K3a: per-leaf GEMM1 (256x768 x 768x4096) + bias + GEGLU (WMMA) ==========
// LDS-staged weight panels (v-half and gate-half, transposed [col][k]),
// A fragment reused across all N-tiles and both halves.
// grid = leaf(32) * colslice(32 x 64 cols) * Mhalf(2) = 2048 blocks
__global__ void k3a_ffn1(const float* __restrict__ slots2, const float* __restrict__ w1,
                         const float* __restrict__ b1, float* __restrict__ actb) {
  const int leaf = blockIdx.x >> 6;            // 0..31 = e1*4+e2
  const int rem  = blockIdx.x & 63;
  const int c0   = (rem >> 1) * 64;            // hidden column block [0,2048)
  const int mh   = rem & 1;                    // M half
  const int e1 = leaf >> 2, e2 = leaf & 3;
  const int lane = threadIdx.x & 31, w = threadIdx.x >> 5;
  const float* W1 = w1 + (size_t)leaf * Dd * (2 * HIDc);
  const float* B1 = b1 + (size_t)leaf * (2 * HIDc);
  float* A_ = actb + (size_t)leaf * GRP * HIDc;
  __shared__ float ldsV[64 * TS];              // 9 KB, transposed [col][k]
  __shared__ float ldsG[64 * TS];              // 9 KB
  const int mt = mh * 8 + w;                   // this wave's M tile
  const int arow = mt * 16 + (lane & 15);
  const float* arowp =
      slots2 + ((((size_t)(arow >> 2) * E1c + e1) * E2c + e2) * Mm + (arow & 3)) * Dd;
  v8f av[4] = {}, ag[4] = {};
  for (int kt = 0; kt < Dd / 32; ++kt) {
    __syncthreads();
    {  // stage 32x64 panels of v-half and gate-half (coalesced reads)
      const int k  = threadIdx.x >> 3;         // 0..31
      const int cg = (threadIdx.x & 7) * 8;    // 0..56
      const float* gv = W1 + (size_t)(kt * 32 + k) * (2 * HIDc) + c0 + cg;
      __builtin_prefetch(gv + (size_t)32 * (2 * HIDc), 0, 0);
#pragma unroll
      for (int j = 0; j < 8; ++j) ldsV[(cg + j) * TS + k] = gv[j];
#pragma unroll
      for (int j = 0; j < 8; ++j) ldsG[(cg + j) * TS + k] = gv[HIDc + j];
    }
    __syncthreads();
    v16bf a = a_frag(arowp + kt * 32, lane);
#pragma unroll
    for (int n = 0; n < 4; ++n) {
      const int cl = n * 16 + (lane & 15);
      const int kb = (lane < 16) ? 0 : 16;
      const float* pv = &ldsV[cl * TS + kb];
      const float* pg = &ldsG[cl * TS + kb];
      v16bf bv, bg;
#pragma unroll
      for (int e = 0; e < 16; ++e) bv[e] = (__bf16)pv[e];
#pragma unroll
      for (int e = 0; e < 16; ++e) bg[e] = (__bf16)pg[e];
      av[n] = wmma_bf16(a, bv, av[n]);
      ag[n] = wmma_bf16(a, bg, ag[n]);
    }
  }
#pragma unroll
  for (int n = 0; n < 4; ++n) {
    const int col = c0 + n * 16 + (lane & 15);
    const float bv = B1[col], bgt = B1[HIDc + col];
    const int r0 = mt * 16 + ((lane < 16) ? 0 : 8);
#pragma unroll
    for (int r = 0; r < 8; ++r) {
      const float vv = av[n][r] + bv;
      const float gg = ag[n][r] + bgt;
      A_[(size_t)(r0 + r) * HIDc + col] = vv * gelu_tanh(gg);
    }
  }
}

// ====== K3b: per-leaf GEMM2 (256x2048 x 2048x768) + bias (WMMA) ================
// LDS-staged transposed weight panel; each weight byte enters the block once.
// grid = leaf(32) * colslice(6 x 128 cols) * Mhalf(2) = 384 blocks
__global__ void k3b_ffn2(const float* __restrict__ actb, const float* __restrict__ w2,
                         const float* __restrict__ b2, float* __restrict__ oexp) {
  const int leaf = blockIdx.x / 12;
  const int rem  = blockIdx.x % 12;
  const int c0   = (rem >> 1) * 128;
  const int mh   = rem & 1;
  const int e1 = leaf >> 2, e2 = leaf & 3;
  const int lane = threadIdx.x & 31, w = threadIdx.x >> 5;
  const float* Aa = actb + (size_t)leaf * GRP * HIDc;
  const float* W2 = w2 + (size_t)leaf * HIDc * Dd;
  const float* B2 = b2 + (size_t)leaf * Dd;
  __shared__ float ldsB[128 * TS];             // 18 KB, transposed [col][k]
  const int mt = mh * 8 + w;
  const int arow = mt * 16 + (lane & 15);
  v8f acc[8] = {};
  for (int kt = 0; kt < HIDc / 32; ++kt) {
    __syncthreads();
    {  // stage 32x128 W2 panel transposed (coalesced reads)
      const int k  = threadIdx.x >> 3;         // 0..31
      const int cg = (threadIdx.x & 7) * 16;   // 0..112
      const float* g = W2 + (size_t)(kt * 32 + k) * Dd + c0 + cg;
      __builtin_prefetch(g + (size_t)32 * Dd, 0, 0);
#pragma unroll
      for (int j = 0; j < 16; ++j) ldsB[(cg + j) * TS + k] = g[j];
    }
    __syncthreads();
    v16bf a = a_frag(Aa + (size_t)arow * HIDc + kt * 32, lane);
#pragma unroll
    for (int n = 0; n < 8; ++n) {
      const float* p = &ldsB[(n * 16 + (lane & 15)) * TS + ((lane < 16) ? 0 : 16)];
      v16bf b;
#pragma unroll
      for (int e = 0; e < 16; ++e) b[e] = (__bf16)p[e];
      acc[n] = wmma_bf16(a, b, acc[n]);
    }
  }
#pragma unroll
  for (int n = 0; n < 8; ++n) {
    const int col = c0 + n * 16 + (lane & 15);
    const float bb = B2[col];
    const int r0 = mt * 16 + ((lane < 16) ? 0 : 8);
#pragma unroll
    for (int r = 0; r < 8; ++r) {
      const int rr = r0 + r;
      oexp[((((size_t)(rr >> 2) * E1c + e1) * E2c + e2) * Mm + (rr & 3)) * Dd + col] =
          acc[n][r] + bb;
    }
  }
}

// ====== K4: out2[s] = sum_p comb2[s,p] * o[p]  (16x16 x 16x768, VALU) ==========
__global__ void k4_comb2(const float* __restrict__ oexp, const float* __restrict__ comb2,
                         float* __restrict__ out2) {
  const int bg = blockIdx.x / E1c, e1 = blockIdx.x % E1c;
  const float* o  = oexp + ((size_t)bg * E1c + e1) * S1c * Dd;
  const float* c2 = comb2 + ((size_t)bg * E1c + e1) * S1c * S1c;
  float* o2 = out2 + ((size_t)bg * P1 + e1 * S1c) * Dd;
  for (int i = threadIdx.x; i < S1c * Dd; i += 256) {
    const int s = i / Dd, d = i % Dd;
    float acc = 0.f;
#pragma unroll
    for (int p = 0; p < S1c; ++p) acc += c2[s * S1c + p] * o[(size_t)p * Dd + d];
    o2[i] = acc;
  }
}

// ====== K5: out = comb1 (256x128) * out2 (128x768) per group (WMMA) ============
__global__ void k5_comb1(const float* __restrict__ comb1, const float* __restrict__ out2,
                         float* __restrict__ out) {
  const int bg = blockIdx.x;
  const float* Aa = comb1 + (size_t)bg * GRP * P1;
  const float* Bm = out2 + (size_t)bg * P1 * Dd;
  float* Co = out + (size_t)bg * GRP * Dd;
  const int lane = threadIdx.x & 31, w = threadIdx.x >> 5;
#pragma unroll 1
  for (int i = 0; i < 96; ++i) {                 // 16x48 = 768 tiles / 8 waves
    const int tidx = w * 96 + i;
    const int mt = tidx / 48, nt = tidx % 48;
    const int row = mt * 16 + (lane & 15);
    const int col = nt * 16 + (lane & 15);
    v8f acc = {};
    for (int kt = 0; kt < P1 / 32; ++kt) {
      v16bf a = a_frag(Aa + (size_t)row * P1 + kt * 32, lane);
      v16bf b = b_frag(Bm + (size_t)(kt * 32 + ((lane < 16) ? 0 : 16)) * Dd + col, Dd);
      acc = wmma_bf16(a, b, acc);
    }
    store_c(Co, Dd, mt * 16, nt * 16, lane, acc);
  }
}

// ======================= host-side launch sequence =============================
extern "C" void kernel_launch(void* const* d_in, const int* in_sizes, int n_in,
                              void* d_out, int out_size, void* d_ws, size_t ws_size,
                              hipStream_t stream) {
  (void)in_sizes; (void)n_in; (void)out_size; (void)ws_size;
  const float* x        = (const float*)d_in[0];
  const float* gamma_in = (const float*)d_in[1];
  const float* slots1   = (const float*)d_in[2];
  const float* gamma2   = (const float*)d_in[3];
  const float* w_slot2  = (const float*)d_in[4];
  const float* w1       = (const float*)d_in[5];
  const float* b1       = (const float*)d_in[6];
  const float* w2       = (const float*)d_in[7];
  const float* b2       = (const float*)d_in[8];
  float* out = (float*)d_out;

  // workspace layout (floats); total ~73.1M floats (~292 MB)
  float* ws = (float*)d_ws;
  size_t off = 0;
  float* xn      = ws + off; off += (size_t)Bb * Nn * Dd;            // 12.58M
  float* phi     = ws + off; off += (size_t)P1 * Dd;                 // 98K
  float* logits1 = ws + off; off += (size_t)BG * GRP * P1;           // 2.10M (-> disp1)
  float* comb1   = ws + off; off += (size_t)BG * GRP * P1;           // 2.10M
  float* slots   = ws + off; off += (size_t)BG * P1 * Dd;            // 6.29M
  float* ynb     = ws + off; off += (size_t)BG * E1c * S1c * Dd;     // 6.29M
  float* meansb  = ws + off; off += (size_t)BG * E1c * Mm * Dd;      // 1.57M
  float* se2raw  = ws + off; off += (size_t)BG * E1c * Mm * E2c * Dd;// 6.29M
  float* comb2   = ws + off; off += (size_t)BG * E1c * S1c * S1c;    // 131K
  float* slots2  = ws + off; off += (size_t)BG * E1c * S1c * Dd;     // 6.29M
  float* actb    = ws + off; off += (size_t)E1c * E2c * GRP * HIDc;  // 16.78M
  float* oexp    = ws + off; off += (size_t)BG * E1c * S1c * Dd;     // 6.29M
  float* out2    = ws + off; off += (size_t)BG * P1 * Dd;            // 6.29M

  k_rmsnorm  <<<Bb * Nn,      256, 0, stream>>>(x, gamma_in, xn);
  k_phinorm  <<<P1,           256, 0, stream>>>(slots1, phi);
  k1a_logits1<<<BG,           256, 0, stream>>>(xn, phi, logits1);
  k1b_comb1  <<<BG * GRP / 8, 256, 0, stream>>>(logits1, comb1);
  k1c_disp1  <<<BG,           128, 0, stream>>>(logits1);
  k1d_slots  <<<BG,           256, 0, stream>>>(logits1, xn, slots);
  k2a_norm   <<<BG * E1c,     256, 0, stream>>>(slots, gamma2, ynb, meansb);
  k2b_se2    <<<E1c * 24,     256, 0, stream>>>(meansb, w_slot2, se2raw);
  k2c_route  <<<BG * E1c,     256, 0, stream>>>(ynb, se2raw, comb2, slots2);
  k3a_ffn1   <<<32 * 64,      256, 0, stream>>>(slots2, w1, b1, actb);
  k3b_ffn2   <<<32 * 12,      256, 0, stream>>>(actb, w2, b2, oexp);
  k4_comb2   <<<BG * E1c,     256, 0, stream>>>(oexp, comb2, out2);
  k5_comb1   <<<BG,           256, 0, stream>>>(comb1, out2, out);
}
